// InfiniteTalkAudioAttention_73890617360956
// MI455X (gfx1250) — matrix operand order repaired and checked by reading, not compiled, not run on hardware
//
#include <hip/hip_runtime.h>
#include <hip/hip_bf16.h>

// ---------------------------------------------------------------------------
// InfiniteTalk audio cross-attention for MI455X (gfx1250, wave32, WMMA).
// Pipeline:
//   1) f32->f16 weight conversion (4x 2048x2048)
//   2) per-stream min/max of ref_attn_maps, then per-token RoPE positions
//   3) K/V projection GEMM (WMMA f16) -> k_raw f32 / vT f16 (transposed)
//   4) K finalize: RMS-norm + RoPE -> k_att[bt,h,a,d] f16
//   5) Q projection GEMM (WMMA f16, A loaded f32+cvt) -> q_raw f16
//   6) Q finalize: RMS-norm + RoPE + head transpose -> q_att[bt,h,Lp,d] f16
//   7) attention: per (bt,h,mtile) wave: WMMA scores -> shfl softmax ->
//      probs through LDS (layout swap) -> WMMA * vT -> attn_out[bt,l,h*d]
//   8) output projection GEMM (WMMA) -> d_out f32
// GEMM: B K-slabs staged into LDS with GLOBAL_LOAD_ASYNC_TO_LDS_B128
// (double buffered, ASYNCcnt/DScnt/barrier protocol); A register double-
// buffered from global so WMMAs cover all load latency.
// ---------------------------------------------------------------------------

typedef _Float16     v16h  __attribute__((ext_vector_type(16)));
typedef float        v8f   __attribute__((ext_vector_type(8)));
typedef float        f32x4 __attribute__((ext_vector_type(4)));
typedef unsigned int u32x4 __attribute__((ext_vector_type(4)));
typedef unsigned int u32x8 __attribute__((ext_vector_type(8)));

#define T_GRID 21
#define L_TOK  1560
#define LP_TOK 1568      // 98 * 16
#define A_TOK  64
#define DMODEL 2048
#define HEADS  16
#define DHEAD  128
#define S_TOK  (T_GRID * L_TOK)      // 32760
#define M_ENC  (T_GRID * A_TOK)      // 1344
#define NHEADS_TOT (T_GRID * HEADS)  // 336
#define LN_THETA_OVER_HALF 0.14391156f  // ln(10000)/64

// ---------------- fragment load helpers -----------------------------------

__device__ __forceinline__ v16h ld16h_32(const _Float16* p) {
  union { u32x8 u; v16h v; } x;
  x.u = *reinterpret_cast<const u32x8*>(p);   // 2x global_load_b128 / ds_load_b128
  return x.v;
}
__device__ __forceinline__ v16h ld16h_2x16(const _Float16* p0, const _Float16* p1) {
  union { u32x4 q[2]; v16h v; } x;
  x.q[0] = *reinterpret_cast<const u32x4*>(p0);
  x.q[1] = *reinterpret_cast<const u32x4*>(p1);
  return x.v;
}
__device__ __forceinline__ v16h cvt16h_f32(const float* p0, const float* p1) {
  f32x4 a = *reinterpret_cast<const f32x4*>(p0);
  f32x4 b = *reinterpret_cast<const f32x4*>(p0 + 4);
  f32x4 c = *reinterpret_cast<const f32x4*>(p1);
  f32x4 d = *reinterpret_cast<const f32x4*>(p1 + 4);
  v16h r;
#pragma unroll
  for (int j = 0; j < 4; ++j) {
    r[j]      = (_Float16)a[j];
    r[j + 4]  = (_Float16)b[j];
    r[j + 8]  = (_Float16)c[j];
    r[j + 12] = (_Float16)d[j];
  }
  return r;
}

template <bool A_F16>
__device__ __forceinline__ v16h loadAfrag(const void* Aptr, int row, int K, int koff) {
  if (A_F16) {
    const _Float16* p = (const _Float16*)Aptr + (size_t)row * K + koff;
    return ld16h_2x16(p, p + 8);
  } else {
    const float* p = (const float*)Aptr + (size_t)row * K + koff;
    return cvt16h_f32(p, p + 8);
  }
}

// ---------------- utility kernels ------------------------------------------

__global__ void __launch_bounds__(256)
cvt_f32_to_f16(const float* __restrict__ src, _Float16* __restrict__ dst, int n) {
  int i = blockIdx.x * 256 + threadIdx.x;
  int stride = gridDim.x * 256;
  for (; i < n; i += stride) dst[i] = (_Float16)src[i];
}

// one block per audio stream: min/max over S tokens
__global__ void __launch_bounds__(256)
stream_minmax(const float* __restrict__ maps, float* __restrict__ mm, int S) {
  __shared__ float smn[256];
  __shared__ float smx[256];
  const float* p = maps + (size_t)blockIdx.x * S;
  float mn = 3.4e38f, mx = -3.4e38f;
  for (int i = threadIdx.x; i < S; i += 256) {
    float v = p[i];
    mn = fminf(mn, v);
    mx = fmaxf(mx, v);
  }
  smn[threadIdx.x] = mn; smx[threadIdx.x] = mx;
  __syncthreads();
  for (int s = 128; s > 0; s >>= 1) {
    if (threadIdx.x < s) {
      smn[threadIdx.x] = fminf(smn[threadIdx.x], smn[threadIdx.x + s]);
      smx[threadIdx.x] = fmaxf(smx[threadIdx.x], smx[threadIdx.x + s]);
    }
    __syncthreads();
  }
  if (threadIdx.x == 0) { mm[blockIdx.x * 2] = smn[0]; mm[blockIdx.x * 2 + 1] = smx[0]; }
}

// per-token RoPE position: normalize winning stream into its label range
__global__ void __launch_bounds__(256)
compute_pos(const float* __restrict__ maps, const float* __restrict__ mm,
            float* __restrict__ pos, int S) {
  int n = blockIdx.x * 256 + threadIdx.x;
  if (n >= S) return;
  float m0 = maps[n], m1 = maps[S + n];
  int   win = (m1 > m0) ? 1 : 0;            // argmax, first wins ties
  float v   = win ? m1 : m0;
  float lo  = win ? 20.0f : 0.0f;           // LABEL_RANGES lows; range = 4 both
  float mn = mm[win * 2], mx = mm[win * 2 + 1];
  pos[n] = (v - mn) / (mx - mn + 1e-8f) * 4.0f + lo;
}

// ---------------- WMMA GEMM: C[M,N] = A[M,K] * W[N,K]^T + bias -------------
// block 128x128 (8 waves, waveM 0..3 x waveN 0..1), wave tile 32x64.
// B K-slabs (128 cols x 64 halves = 16KB) are async-DMA'd to LDS
// (GLOBAL_LOAD_ASYNC_TO_LDS_B128, double buffered) and shared by all waves;
// fragments then come from LDS as ds_load_b128 with k = 16*half + s per lane (same k order as A).
// Column stride padded to 72 halves (144B) to spread LDS banks.

enum { STORE_F16 = 0, STORE_F32 = 1, STORE_VT = 2 };

#define BSTRIDE 72

template <bool A_F16, int EPI>
__global__ void __launch_bounds__(256)
gemm_wmma(const void* __restrict__ Aptr, const _Float16* __restrict__ Bw,
          const float* __restrict__ bias, void* __restrict__ Cptr,
          int M, int N, int K) {
  __shared__ _Float16 Bsh[2][128 * BSTRIDE];   // 2 x 18KB
  const int lane = threadIdx.x & 31;
  const int wid  = threadIdx.x >> 5;
  const int waveM = wid & 3, waveN = wid >> 2;
  const int rowBase  = blockIdx.x * 128 + waveM * 32;
  const int blockCol = blockIdx.y * 128;
  const int lr = lane & 15, lh = lane >> 4;

  v8f acc[2][4];
#pragma unroll
  for (int i = 0; i < 2; ++i)
#pragma unroll
    for (int j = 0; j < 4; ++j) acc[i][j] = {};

  int rowA[2];
#pragma unroll
  for (int i = 0; i < 2; ++i) {
    int r = rowBase + i * 16 + lr;
    rowA[i] = (r < M) ? r : (M - 1);        // clamp edge-tile loads
  }

  // async slab copy: 1024 16B chunks, 4 per thread. chunk q -> col q/8,
  // k-halves (q%8)*8. LDS byte address = low 32 bits of generic LDS pointer.
  const int q0 = threadIdx.x * 4;
  auto issue_slab = [&](int buf, int k0) {
#pragma unroll
    for (int i = 0; i < 4; ++i) {
      int q  = q0 + i;
      int c  = q >> 3;
      int kk = (q & 7) * 8;
      unsigned long long ga =
          (unsigned long long)(uintptr_t)(Bw + (size_t)(blockCol + c) * K + k0 + kk);
      unsigned int la = (unsigned int)(uintptr_t)&Bsh[buf][c * BSTRIDE + kk];
      __asm__ volatile("global_load_async_to_lds_b128 %0, %1, off"
                       :: "v"(la), "v"(ga) : "memory");
    }
  };

  issue_slab(0, 0);

  v16h aCur[2][2], aNxt[2][2];
#pragma unroll
  for (int c = 0; c < 2; ++c)
#pragma unroll
    for (int i = 0; i < 2; ++i)
      aCur[c][i] = loadAfrag<A_F16>(Aptr, rowA[i], K, c * 32 + lh * 16);

  int ibuf = 0;
  for (int k0 = 0; k0 < K; k0 += 64) {   // K % 64 == 0 (K = 2048)
    // my async chunks for slab(ibuf) landed; my LDS reads of slab(ibuf^1)
    // from last iteration landed; then rendezvous so everyone agrees.
    __asm__ volatile("s_wait_asynccnt 0x0" ::: "memory");
    __asm__ volatile("s_wait_dscnt 0x0" ::: "memory");
    __syncthreads();

    const bool more = (k0 + 64) < K;
    if (more) issue_slab(ibuf ^ 1, k0 + 64);   // overlap DMA with WMMAs
    if (more) {
#pragma unroll
      for (int c = 0; c < 2; ++c)
#pragma unroll
        for (int i = 0; i < 2; ++i)
          aNxt[c][i] = loadAfrag<A_F16>(Aptr, rowA[i], K, k0 + 64 + c * 32 + lh * 16);
    }

#pragma unroll
    for (int c = 0; c < 2; ++c) {
      v16h b[4];
#pragma unroll
      for (int j = 0; j < 4; ++j)
        b[j] = ld16h_32(&Bsh[ibuf][(waveN * 64 + j * 16 + lr) * BSTRIDE + c * 32 + lh * 16]);
#pragma unroll
      for (int i = 0; i < 2; ++i)
#pragma unroll
        for (int j = 0; j < 4; ++j)
          acc[i][j] = __builtin_amdgcn_wmma_f32_16x16x32_f16(
              false, aCur[c][i], false, b[j], (short)0, acc[i][j], false, false);
    }

#pragma unroll
    for (int c = 0; c < 2; ++c)
#pragma unroll
      for (int i = 0; i < 2; ++i)
        aCur[c][i] = aNxt[c][i];
    ibuf ^= 1;
  }

#pragma unroll
  for (int j = 0; j < 4; ++j) {
    const int col = blockCol + waveN * 64 + j * 16 + lr;
    const float bv = bias[col];
#pragma unroll
    for (int i = 0; i < 2; ++i) {
#pragma unroll
      for (int r = 0; r < 8; ++r) {
        int row = rowBase + i * 16 + r + lh * 8;   // C layout: M = r + 8*half
        if (row < M) {
          float val = acc[i][j][r] + bv;
          if (EPI == STORE_F16) {
            ((_Float16*)Cptr)[(size_t)row * N + col] = (_Float16)val;
          } else if (EPI == STORE_F32) {
            ((float*)Cptr)[(size_t)row * N + col] = val;
          } else {  // STORE_VT: row=(bt,a), col=(h,d) -> vT[bt][h][d][a]
            int bt = row >> 6, aa = row & 63, hh = col >> 7, dd = col & 127;
            ((_Float16*)Cptr)[(((size_t)bt * HEADS + hh) * DHEAD + dd) * A_TOK + aa] =
                (_Float16)val;
          }
        }
      }
    }
  }
}

// ---------------- RMS-norm + RoPE finalize passes --------------------------

__device__ __forceinline__ float block_sumsq_reduce(float* red, float v) {
  red[threadIdx.x] = v;
  __syncthreads();
  for (int s = 128; s > 0; s >>= 1) {
    if (threadIdx.x < s) red[threadIdx.x] += red[threadIdx.x + s];
    __syncthreads();
  }
  return red[0];
}

// one block per (bt, l) padded row; thread t owns cols t*8..t*8+7
__global__ void __launch_bounds__(256)
q_finalize(const _Float16* __restrict__ q_raw, const float* __restrict__ pos_ids,
           const float* __restrict__ norm_w, _Float16* __restrict__ q_att) {
  __shared__ float red[256];
  __shared__ float rowf[DMODEL];
  const int row = blockIdx.x;              // 0 .. 21*1568-1
  const int bt = row / LP_TOK, l = row - bt * LP_TOK;
  const int t = threadIdx.x;
  const int h = t >> 4, dbase = (t & 15) * 8;
  const size_t obase = (((size_t)bt * HEADS + h) * LP_TOK + l) * DHEAD;

  if (l >= L_TOK) {                        // zero pad rows for clean WMMA tiles
#pragma unroll
    for (int j = 0; j < 8; ++j) q_att[obase + dbase + j] = (_Float16)0.0f;
    return;
  }
  const size_t g = (size_t)bt * L_TOK + l;
  const _Float16* qr = q_raw + g * DMODEL;
  float ss = 0.0f;
#pragma unroll
  for (int j = 0; j < 8; ++j) {
    float v = (float)qr[t * 8 + j];
    rowf[t * 8 + j] = v;
    ss += v * v;
  }
  float tot = block_sumsq_reduce(red, ss);
  float rinv = __frsqrt_rn(tot * (1.0f / (float)DMODEL) + 1e-5f);
  float pos = pos_ids[g];
#pragma unroll
  for (int j = 0; j < 8; ++j) {
    int d = dbase + j;
    int col = h * DHEAD + d;
    int pcol = h * DHEAD + ((d < 64) ? d + 64 : d - 64);
    float f = pos * __expf(-LN_THETA_OVER_HALF * (float)(d & 63));
    float x1 = rowf[col]  * rinv * norm_w[col];
    float x2 = rowf[pcol] * rinv * norm_w[pcol];
    float val = x1 * __cosf(f) + ((d < 64) ? -x2 : x2) * __sinf(f);
    q_att[obase + d] = (_Float16)val;
  }
}

// one block per encoder row m = bt*64 + a
__global__ void __launch_bounds__(256)
k_finalize(const float* __restrict__ k_raw, const float* __restrict__ norm_w,
           _Float16* __restrict__ k_att) {
  __shared__ float red[256];
  __shared__ float rowf[DMODEL];
  const int m = blockIdx.x;
  const int bt = m >> 6, aa = m & 63;
  const int t = threadIdx.x;
  const float* kr = k_raw + (size_t)m * DMODEL;
  float ss = 0.0f;
#pragma unroll
  for (int j = 0; j < 8; ++j) {
    float v = kr[t * 8 + j];
    rowf[t * 8 + j] = v;
    ss += v * v;
  }
  float tot = block_sumsq_reduce(red, ss);
  float rinv = __frsqrt_rn(tot * (1.0f / (float)DMODEL) + 1e-5f);
  // enc positions: label-range midpoints repeated per half-frame (chunk=32)
  float pos = (aa < 32) ? 2.0f : 22.0f;
  const int h = t >> 4, dbase = (t & 15) * 8;
#pragma unroll
  for (int j = 0; j < 8; ++j) {
    int d = dbase + j;
    int col = h * DHEAD + d;
    int pcol = h * DHEAD + ((d < 64) ? d + 64 : d - 64);
    float f = pos * __expf(-LN_THETA_OVER_HALF * (float)(d & 63));
    float x1 = rowf[col]  * rinv * norm_w[col];
    float x2 = rowf[pcol] * rinv * norm_w[pcol];
    float val = x1 * __cosf(f) + ((d < 64) ? -x2 : x2) * __sinf(f);
    k_att[(((size_t)bt * HEADS + h) * A_TOK + aa) * DHEAD + d] = (_Float16)val;
  }
}

// ---------------- fused cross-attention (one wave per 16-row tile) ---------
// grid: x = bt*16+h (336), y = L-tile (98).  A=64 keys, d=128.
__global__ void __launch_bounds__(32)
attention_wmma(const _Float16* __restrict__ q_att, const _Float16* __restrict__ k_att,
               const _Float16* __restrict__ vT, _Float16* __restrict__ attn_out) {
  __shared__ _Float16 P[16 * 64];          // probabilities, row-major [m][a]
  const int bh = blockIdx.x;
  const int bt = bh >> 4, h = bh & 15;
  const int mt = blockIdx.y;
  const int lane = threadIdx.x;
  const int lr = lane & 15, lh = lane >> 4;

  const _Float16* qp = q_att + ((size_t)bh * LP_TOK + mt * 16) * DHEAD;
  const _Float16* kp = k_att + (size_t)bh * A_TOK * DHEAD;
  const _Float16* vp = vT    + (size_t)bh * DHEAD * A_TOK;

  // Q fragments: 4 K-chunks over d=128
  v16h aq[4];
#pragma unroll
  for (int c = 0; c < 4; ++c) {
    const _Float16* p = qp + (size_t)lr * DHEAD + c * 32 + lh * 16;
    aq[c] = ld16h_2x16(p, p + 8);
  }

  // scores: S[m, a] = q . k  — load all 4 B fragments of a tile as one clause,
  // then run 4 WMMAs so a single wait covers the whole tile.
  v8f s[4];
#pragma unroll
  for (int n = 0; n < 4; ++n) {
    v16h b[4];
#pragma unroll
    for (int c = 0; c < 4; ++c)
      b[c] = ld16h_32(kp + (size_t)(n * 16 + lr) * DHEAD + c * 32 + lh * 16);
    s[n] = {};
#pragma unroll
    for (int c = 0; c < 4; ++c)
      s[n] = __builtin_amdgcn_wmma_f32_16x16x32_f16(false, aq[c], false, b[c],
                                                    (short)0, s[n], false, false);
  }

  // softmax over 64 keys; row m = r + 8*lh lives in one 16-lane half-group
  const float sc = 0.08838834764831844f;   // 1/sqrt(128)
#pragma unroll
  for (int r = 0; r < 8; ++r) {
    float m = fmaxf(fmaxf(s[0][r], s[1][r]), fmaxf(s[2][r], s[3][r]));
#pragma unroll
    for (int msk = 1; msk < 16; msk <<= 1) m = fmaxf(m, __shfl_xor(m, msk, 32));
    float tsum = 0.0f;
#pragma unroll
    for (int n = 0; n < 4; ++n) {
      float e = __expf(sc * (s[n][r] - m));
      s[n][r] = e;
      tsum += e;
    }
#pragma unroll
    for (int msk = 1; msk < 16; msk <<= 1) tsum += __shfl_xor(tsum, msk, 32);
    float inv = 1.0f / tsum;
#pragma unroll
    for (int n = 0; n < 4; ++n)
      P[(r + lh * 8) * 64 + n * 16 + lr] = (_Float16)(s[n][r] * inv);
  }
  // single wave: LDS pipe is in-order, but be explicit about the DS hazard
  __asm__ volatile("s_wait_dscnt 0x0" ::: "memory");

  // probs back as A fragments (K = 64 -> 2 chunks)
  v16h ap[2];
#pragma unroll
  for (int c2 = 0; c2 < 2; ++c2) {
    const _Float16* p = &P[lr * 64 + c2 * 32 + lh * 16];
    ap[c2] = ld16h_2x16(p, p + 8);
  }

  // out[m, d] = P . v   (B fragments from vT[d][a]); store to [bt, l, h*128+d]
#pragma unroll
  for (int nd = 0; nd < 8; ++nd) {
    v16h b[2];
#pragma unroll
    for (int c2 = 0; c2 < 2; ++c2)
      b[c2] = ld16h_32(vp + (size_t)(nd * 16 + lr) * A_TOK + c2 * 32 + lh * 16);
    v8f o = {};
#pragma unroll
    for (int c2 = 0; c2 < 2; ++c2)
      o = __builtin_amdgcn_wmma_f32_16x16x32_f16(false, ap[c2], false, b[c2],
                                                 (short)0, o, false, false);
#pragma unroll
    for (int r = 0; r < 8; ++r) {
      int row = mt * 16 + r + lh * 8;
      if (row < L_TOK)
        attn_out[((size_t)bt * L_TOK + row) * DMODEL + h * DHEAD + nd * 16 + lr] =
            (_Float16)o[r];
    }
  }
}

// ---------------- host-side launch ------------------------------------------

extern "C" void kernel_launch(void* const* d_in, const int* in_sizes, int n_in,
                              void* d_out, int out_size, void* d_ws, size_t ws_size,
                              hipStream_t stream) {
  (void)in_sizes; (void)n_in; (void)out_size; (void)ws_size;

  const float* hidden = (const float*)d_in[0];   // [1, 32760, 2048]
  const float* enc    = (const float*)d_in[1];   // [21, 64, 2048]
  const float* maps   = (const float*)d_in[2];   // [2, 32760]
  const float* Wq     = (const float*)d_in[3];
  const float* bq     = (const float*)d_in[4];
  const float* Wk     = (const float*)d_in[5];
  const float* bk     = (const float*)d_in[6];
  const float* Wv     = (const float*)d_in[7];
  const float* bv     = (const float*)d_in[8];
  const float* Wo     = (const float*)d_in[9];
  const float* bo     = (const float*)d_in[10];
  const float* nqw    = (const float*)d_in[11];
  const float* nkw    = (const float*)d_in[12];

  // ---- workspace layout (256B aligned), with sequential aliasing ----
  char* ws = (char*)d_ws;
  size_t off = 0;
  auto take = [&](size_t bytes) { size_t o = off; off += (bytes + 255) & ~(size_t)255; return o; };

  const size_t WBYTES = (size_t)DMODEL * DMODEL * 2;           // 8 MB each
  size_t oWq = take(WBYTES), oWk = take(WBYTES), oWv = take(WBYTES), oWo = take(WBYTES);
  size_t oBufA = take((size_t)S_TOK * DMODEL * 2);             // q_raw, later attn_out
  size_t oBufB = take((size_t)NHEADS_TOT * LP_TOK * DHEAD * 2);// q_att; k_raw aliased at start
  size_t oKatt = take((size_t)NHEADS_TOT * A_TOK * DHEAD * 2);
  size_t oVt   = take((size_t)NHEADS_TOT * DHEAD * A_TOK * 2);
  size_t oMM   = take(16);
  size_t oPos  = take((size_t)S_TOK * 4);

  _Float16* Wq16 = (_Float16*)(ws + oWq);
  _Float16* Wk16 = (_Float16*)(ws + oWk);
  _Float16* Wv16 = (_Float16*)(ws + oWv);
  _Float16* Wo16 = (_Float16*)(ws + oWo);
  _Float16* q_raw    = (_Float16*)(ws + oBufA);   // written step 5, read step 6
  _Float16* attn_buf = (_Float16*)(ws + oBufA);   // written step 7, read step 8
  float*    k_raw    = (float*)   (ws + oBufB);   // written step 3, read step 4
  _Float16* q_att    = (_Float16*)(ws + oBufB);   // written step 6 (after k_raw dead)
  _Float16* k_att    = (_Float16*)(ws + oKatt);
  _Float16* vT       = (_Float16*)(ws + oVt);
  float*    mm       = (float*)   (ws + oMM);
  float*    pos      = (float*)   (ws + oPos);

  const int WN = DMODEL * DMODEL;

  // 1) weights -> f16
  cvt_f32_to_f16<<<2048, 256, 0, stream>>>(Wq, Wq16, WN);
  cvt_f32_to_f16<<<2048, 256, 0, stream>>>(Wk, Wk16, WN);
  cvt_f32_to_f16<<<2048, 256, 0, stream>>>(Wv, Wv16, WN);
  cvt_f32_to_f16<<<2048, 256, 0, stream>>>(Wo, Wo16, WN);

  // 2) RoPE positions
  stream_minmax<<<2, 256, 0, stream>>>(maps, mm, S_TOK);
  compute_pos<<<(S_TOK + 255) / 256, 256, 0, stream>>>(maps, mm, pos, S_TOK);

  // 3) K/V projections (M=1344)
  dim3 gKV((M_ENC + 127) / 128, DMODEL / 128);
  gemm_wmma<false, STORE_F32><<<gKV, 256, 0, stream>>>(enc, Wk16, bk, k_raw,
                                                       M_ENC, DMODEL, DMODEL);
  gemm_wmma<false, STORE_VT><<<gKV, 256, 0, stream>>>(enc, Wv16, bv, vT,
                                                      M_ENC, DMODEL, DMODEL);
  // 4) K finalize (norm + RoPE + head split)
  k_finalize<<<M_ENC, 256, 0, stream>>>(k_raw, nkw, k_att);

  // 5) Q projection (M=32760, A loaded as f32 and converted in-register)
  dim3 gQ((S_TOK + 127) / 128, DMODEL / 128);
  gemm_wmma<false, STORE_F16><<<gQ, 256, 0, stream>>>(hidden, Wq16, bq, q_raw,
                                                      S_TOK, DMODEL, DMODEL);
  // 6) Q finalize (norm + RoPE + transpose into padded attention layout)
  q_finalize<<<T_GRID * LP_TOK, 256, 0, stream>>>(q_raw, pos, nqw, q_att);

  // 7) fused attention
  dim3 gAtt(NHEADS_TOT, LP_TOK / 16);
  attention_wmma<<<gAtt, 32, 0, stream>>>(q_att, k_att, vT, attn_buf);

  // 8) output projection -> f32 result
  gemm_wmma<true, STORE_F32><<<gQ, 256, 0, stream>>>(attn_buf, Wo16, bo, d_out,
                                                     S_TOK, DMODEL, DMODEL);
}
